// LinearDiffusionModel_29463475651174
// MI455X (gfx1250) — matrix operand order repaired and checked by reading, not compiled, not run on hardware
//
#include <hip/hip_runtime.h>
#include <hip/hip_bf16.h>

typedef __attribute__((ext_vector_type(16))) __bf16 v16bf;
typedef __attribute__((ext_vector_type(8)))  float  v8f;

#define D 128

// ---------------- degree / normalization ----------------
__global__ void deg_kernel(const int* __restrict__ dst, float* __restrict__ deg, int E) {
    int e = blockIdx.x * blockDim.x + threadIdx.x;
    if (e < E) atomicAdd(&deg[dst[e]], 1.0f);
}

__global__ void invdeg_kernel(float* __restrict__ deg, int N) {
    int n = blockIdx.x * blockDim.x + threadIdx.x;
    if (n < N) deg[n] = 1.0f / fmaxf(deg[n], 1.0f);
}

// ---------------- pack weights into WMMA B-fragment order (bf16) ----------------
// Wt index = ((ntile*4 + kstep)*32 + lane)*16 + e
//   lane<16 : column n = ntile*16 + lane,     K = kstep*32 + e        (e=0..15)
//   lane>=16: column n = ntile*16 + lane-16,  K = kstep*32 + 16 + e
__global__ void pack_w(const float* __restrict__ W, __bf16* __restrict__ Wt) {
    int idx  = blockIdx.x * blockDim.x + threadIdx.x;   // 0..16383
    int e    = idx & 15;
    int lane = (idx >> 4) & 31;
    int s    = (idx >> 9) & 3;
    int t    = (idx >> 11) & 7;
    int n    = t * 16 + (lane & 15);
    int k    = s * 32 + (lane >> 4) * 16 + e;
    Wt[idx]  = (__bf16)W[k * D + n];
}

// ---------------- diffusion: scatter one hop (unnormalized) ----------------
// agg[dst] += term[src].  Normalization by invdeg[dst] is factored out and
// applied per-node in update_kernel (w depends only on dst).
// One wave per edge; each lane owns 4 consecutive features. term + agg are
// L2-resident (51MB each vs 192MB L2), so atomics resolve in L2.
__global__ void scatter_kernel(const int* __restrict__ src, const int* __restrict__ dst,
                               const float* __restrict__ term, float* __restrict__ agg,
                               int E) {
    unsigned t    = blockIdx.x * blockDim.x + threadIdx.x;
    unsigned e    = t >> 5;
    unsigned lane = t & 31u;
    if (e >= (unsigned)E) return;
    int s = src[e];
    int d = dst[e];
    const float4 v = *(const float4*)(term + (size_t)s * D + lane * 4);
    float* a = agg + (size_t)d * D + lane * 4;
    atomicAdd(a + 0, v.x);
    atomicAdd(a + 1, v.y);
    atomicAdd(a + 2, v.z);
    atomicAdd(a + 3, v.w);
}

// t = agg * invdeg[node] / i ; term = t ; acc += t ; agg = 0 (ready for next hop)
// float4-vectorized: one thread per 4 features, 32 float4 per node.
__global__ void update_kernel(float* __restrict__ agg, const float* __restrict__ invdeg,
                              float inv_i, float* __restrict__ term,
                              float* __restrict__ acc, unsigned total4) {
    unsigned i = blockIdx.x * blockDim.x + threadIdx.x;
    if (i >= total4) return;
    const float w = invdeg[i >> 5] * inv_i;   // node = i / (128/4)
    float4 g = ((const float4*)agg)[i];
    float4 t = {g.x * w, g.y * w, g.z * w, g.w * w};
    ((float4*)term)[i] = t;
    float4 a = ((const float4*)acc)[i];
    a.x += t.x; a.y += t.y; a.z += t.z; a.w += t.w;
    ((float4*)acc)[i] = a;
    ((float4*)agg)[i] = float4{0.f, 0.f, 0.f, 0.f};
}

// ---------------- GEMM: [M x 128] * [128 x 128] (+bias, optional SiLU) ----------------
// 256 threads = 8 waves; block owns a 16-row M-tile, wave w owns N-tile w.
// K=128 -> 4 x V_WMMA_F32_16X16X32_BF16; B from pre-packed bf16 fragments.
__global__ __launch_bounds__(256)
void gemm_wmma(const float* __restrict__ A, const __bf16* __restrict__ Wt,
               const float* __restrict__ bias, float* __restrict__ out,
               float* __restrict__ out2, int apply_silu) {
    const int lane = threadIdx.x & 31;
    const int wave = threadIdx.x >> 5;
    const int half = lane >> 4;
    const int l15  = lane & 15;
    const size_t m_base = (size_t)blockIdx.x * 16;
    const int n_base = wave * 16;

    const int m = (int)m_base + l15;  // A row this lane supplies
    const int n = n_base + l15;       // B/C/D column this lane owns

    v8f c = {0.f, 0.f, 0.f, 0.f, 0.f, 0.f, 0.f, 0.f};
    const float* arow = A + (size_t)m * D;
    const __bf16* wfrag = Wt + (((size_t)wave * 4) * 32 + lane) * 16;

    #pragma unroll
    for (int s = 0; s < 4; ++s) {
        const int k0 = s * 32;
        v16bf a;
        // A 16x32 bf16 fragment: lanes 0-15 -> K = k0+{0..15};
        // lanes 16-31 -> K = k0+{16..31}
        const int ka = k0 + half * 16;
        #pragma unroll
        for (int j = 0; j < 8; ++j) {
            a[j]     = (__bf16)arow[ka + j];
            a[8 + j] = (__bf16)arow[ka + 8 + j];
        }
        // B fragment: pre-packed, contiguous 32B per lane
        v16bf b = *(const v16bf*)(wfrag + (size_t)s * 32 * 16);

        c = __builtin_amdgcn_wmma_f32_16x16x32_bf16(
                /*neg_a=*/false, a, /*neg_b=*/false, b,
                /*c_mod=*/(short)0, c, /*reuse_a=*/false, /*reuse_b=*/false);
    }

    // C/D layout: VGPR j -> row (m_base + j + 8*half), col = n
    const float bn = bias[n];
    #pragma unroll
    for (int j = 0; j < 8; ++j) {
        int mr = (int)m_base + j + half * 8;
        float v = c[j] + bn;
        if (apply_silu) v = v * (1.0f / (1.0f + __expf(-v)));
        out[(size_t)mr * D + n] = v;
        if (out2) out2[(size_t)mr * D + n] = v;
    }
}

// ---------------- launcher ----------------
extern "C" void kernel_launch(void* const* d_in, const int* in_sizes, int n_in,
                              void* d_out, int out_size, void* d_ws, size_t ws_size,
                              hipStream_t stream) {
    const float* h   = (const float*)d_in[0];
    const int*   ei  = (const int*)d_in[1];
    const float* Win = (const float*)d_in[2];
    const float* bin = (const float*)d_in[3];
    const float* W1  = (const float*)d_in[4];
    const float* b1  = (const float*)d_in[5];
    const float* W2  = (const float*)d_in[6];
    const float* b2  = (const float*)d_in[7];

    const int N = in_sizes[0] / D;     // 100000 (divisible by 16)
    const int E = in_sizes[1] / 2;     // 1600000
    const int* src = ei;
    const int* dst = ei + E;

    char* ws = (char*)d_ws;
    const size_t featBytes = (size_t)N * D * sizeof(float);
    float*  term = (float*)ws;   ws += featBytes;
    float*  acc  = (float*)ws;   ws += featBytes;
    float*  agg  = (float*)ws;   ws += featBytes;
    float*  deg  = (float*)ws;   ws += (size_t)N * sizeof(float);
    __bf16* Wt0  = (__bf16*)ws;  ws += 16384 * sizeof(__bf16);
    __bf16* Wt1  = (__bf16*)ws;  ws += 16384 * sizeof(__bf16);
    __bf16* Wt2  = (__bf16*)ws;  ws += 16384 * sizeof(__bf16);

    // pack the three weight matrices into bf16 B-fragment layout
    pack_w<<<64, 256, 0, stream>>>(Win, Wt0);
    pack_w<<<64, 256, 0, stream>>>(W1,  Wt1);
    pack_w<<<64, 256, 0, stream>>>(W2,  Wt2);

    // in-degree -> inverse (stored in place)
    hipMemsetAsync(deg, 0, (size_t)N * sizeof(float), stream);
    deg_kernel<<<(E + 255) / 256, 256, 0, stream>>>(dst, deg, E);
    invdeg_kernel<<<(N + 255) / 256, 256, 0, stream>>>(deg, N);

    const int mtiles = N / 16;
    // x = silu(h @ W_in + b_in); seed term = acc = x
    gemm_wmma<<<mtiles, 256, 0, stream>>>(h, Wt0, bin, term, acc, 1);

    // agg zeroed once; update_kernel re-zeroes it each hop after consuming it
    hipMemsetAsync(agg, 0, featBytes, stream);

    const unsigned total4 = (unsigned)N * (D / 4);
    const int upd_blocks = (int)((total4 + 255) / 256);
    const long long sthreads = (long long)E * 32;
    const int sc_blocks = (int)((sthreads + 255) / 256);

    for (int i = 1; i <= 6; ++i) {
        scatter_kernel<<<sc_blocks, 256, 0, stream>>>(src, dst, term, agg, E);
        update_kernel<<<upd_blocks, 256, 0, stream>>>(agg, deg, 1.0f / (float)i,
                                                      term, acc, total4);
    }

    // y = silu(acc @ W1 + b1)   (reuse `term` as y)
    gemm_wmma<<<mtiles, 256, 0, stream>>>(acc, Wt1, b1, term, nullptr, 1);
    // out = y @ W2 + b2
    gemm_wmma<<<mtiles, 256, 0, stream>>>(term, Wt2, b2, (float*)d_out, nullptr, 0);
}
